// ExperimentalClusteringLoss_51230369907022
// MI455X (gfx1250) — matrix operand order repaired and checked by reading, not compiled, not run on hardware
//
#include <hip/hip_runtime.h>
#include <math.h>

// ---------------------------------------------------------------------------
// ExperimentalClusteringLoss for MI455X (gfx1250, wave32, WMMA)
//   pass 0: zero scalar accumulators in workspace
//   pass 1: row softmax (D=128) for both streams; self-sim dot accumulation;
//           write (softmax^2 * 256) as f16, TRANSPOSED (feature-major) so the
//           Gram GEMM reads both operands contiguously along K.
//   pass 2: per (ci<cj, stream) cluster pair: 128x128x512 GEMM via
//           v_wmma_f32_16x16x32_f16 with ASYNC global->LDS staging,
//           fused v_sqrt_f32 + reduction (G never hits HBM)
//   pass 3: combine scalars.
// ---------------------------------------------------------------------------

#define NC      64
#define DD      128
#define BB      512
#define KTOT    512                 // Gram K dimension = batch
#define NPAIRS  2016                // 64*63/2
#define QSCALE  256.0f              // f16 dynamic-range guard for softmax^2

typedef _Float16 v16h __attribute__((ext_vector_type(16)));
typedef _Float16 h8   __attribute__((ext_vector_type(8)));
typedef float    v8f  __attribute__((ext_vector_type(8)));
typedef int      vi4  __attribute__((vector_size(16)));   // b128 payload type

#if defined(__gfx1250__) && __has_builtin(__builtin_amdgcn_global_load_async_to_lds_b128)
#define USE_ASYNC_LDS 1
#else
#define USE_ASYNC_LDS 0
#endif

#if USE_ASYNC_LDS
// global (AS1) -> LDS (AS3) 16-byte async copy, no VGPR data path (ASYNCcnt).
// Builtin expects v4i-typed pointers (per hipcc diagnostic, round 2).
__device__ __forceinline__ void async_copy16(const void* gsrc, void* ldst) {
    auto gp = (__attribute__((address_space(1))) vi4*)(uintptr_t)gsrc;
    auto lp = (__attribute__((address_space(3))) vi4*)(unsigned)(uintptr_t)ldst;
    __builtin_amdgcn_global_load_async_to_lds_b128(gp, lp, 0, 0);
}
__device__ __forceinline__ void wait_async0() {
#if __has_builtin(__builtin_amdgcn_s_wait_asynccnt)
    __builtin_amdgcn_s_wait_asynccnt(0);
#else
    asm volatile("s_wait_asynccnt 0x0" ::: "memory");
#endif
}
#endif

__global__ void eccl_zero_kernel(float* __restrict__ acc) {
    if (threadIdx.x < 2) acc[threadIdx.x] = 0.0f;
}

// ---------------------------------------------------------------------------
// Pass 1: one wave per (b, c) pair, handling stream A and stream B rows.
// Block = 8 waves = 8 consecutive batch indices for one cluster c.
// LDS transpose so the f16 outputs are written as coalesced 16B runs.
// ---------------------------------------------------------------------------
__global__ __launch_bounds__(256) void eccl_softmax_kernel(
    const float* __restrict__ emb,
    _Float16* __restrict__ XaT, _Float16* __restrict__ XbT,
    float* __restrict__ acc)
{
    __shared__ __align__(16) _Float16 qs[2][DD][8];   // [stream][feature][b-in-tile]

    const int c    = blockIdx.x >> 6;          // cluster
    const int b0   = (blockIdx.x & 63) * 8;    // batch tile origin
    const int w    = threadIdx.x >> 5;         // wave id -> batch offset
    const int lane = threadIdx.x & 31;
    const int b    = b0 + w;

    const float* pa = emb + (size_t)b        * (NC * DD) + c * DD + lane * 4;
    const float* pb = emb + (size_t)(BB + b) * (NC * DD) + c * DD + lane * 4;
    float4 va = *(const float4*)pa;
    float4 vb = *(const float4*)pb;

    // ---- softmax, stream A ----
    float ma = fmaxf(fmaxf(va.x, va.y), fmaxf(va.z, va.w));
    #pragma unroll
    for (int o = 16; o; o >>= 1) ma = fmaxf(ma, __shfl_xor(ma, o, 32));
    float a0 = __expf(va.x - ma), a1 = __expf(va.y - ma);
    float a2 = __expf(va.z - ma), a3 = __expf(va.w - ma);
    float sa = a0 + a1 + a2 + a3;
    #pragma unroll
    for (int o = 16; o; o >>= 1) sa += __shfl_xor(sa, o, 32);
    float ra = 1.0f / sa;
    a0 *= ra; a1 *= ra; a2 *= ra; a3 *= ra;

    // ---- softmax, stream B ----
    float mb = fmaxf(fmaxf(vb.x, vb.y), fmaxf(vb.z, vb.w));
    #pragma unroll
    for (int o = 16; o; o >>= 1) mb = fmaxf(mb, __shfl_xor(mb, o, 32));
    float b0e = __expf(vb.x - mb), b1e = __expf(vb.y - mb);
    float b2e = __expf(vb.z - mb), b3e = __expf(vb.w - mb);
    float sb = b0e + b1e + b2e + b3e;
    #pragma unroll
    for (int o = 16; o; o >>= 1) sb += __shfl_xor(sb, o, 32);
    float rb = 1.0f / sb;
    b0e *= rb; b1e *= rb; b2e *= rb; b3e *= rb;

    // ---- self-similarity contribution ----
    float dot = a0 * b0e + a1 * b1e + a2 * b2e + a3 * b3e;
    #pragma unroll
    for (int o = 16; o; o >>= 1) dot += __shfl_xor(dot, o, 32);
    if (lane == 0) atomicAdd(&acc[0], dot);

    // ---- stage scaled squares into LDS (transpose) ----
    const int d = lane * 4;
    qs[0][d + 0][w] = (_Float16)(a0 * a0 * QSCALE);
    qs[0][d + 1][w] = (_Float16)(a1 * a1 * QSCALE);
    qs[0][d + 2][w] = (_Float16)(a2 * a2 * QSCALE);
    qs[0][d + 3][w] = (_Float16)(a3 * a3 * QSCALE);
    qs[1][d + 0][w] = (_Float16)(b0e * b0e * QSCALE);
    qs[1][d + 1][w] = (_Float16)(b1e * b1e * QSCALE);
    qs[1][d + 2][w] = (_Float16)(b2e * b2e * QSCALE);
    qs[1][d + 3][w] = (_Float16)(b3e * b3e * QSCALE);
    __syncthreads();

    // ---- coalesced transposed write: thread -> (stream, feature), 8 f16 = 16B ----
    const int s  = threadIdx.x >> 7;         // 0 or 1
    const int fd = threadIdx.x & 127;        // feature within cluster
    _Float16* dst = (s ? XbT : XaT) + ((size_t)c * DD + fd) * KTOT + b0;
    *(uint4*)dst = *(const uint4*)&qs[s][fd][0];
}

// ---------------------------------------------------------------------------
// Pass 2: one block per (stream, ci<cj) cluster pair.
// 256 threads = 8 waves in a 4(M) x 2(N) grid; each wave owns a 32x64 patch
// (2x4 WMMA tiles). Fused sqrt + full reduction -> one atomicAdd per block.
// ---------------------------------------------------------------------------
__global__ __launch_bounds__(256) void eccl_gram_kernel(
    const _Float16* __restrict__ XaT, const _Float16* __restrict__ XbT,
    float* __restrict__ acc)
{
    __shared__ __align__(16) _Float16 Asl[128][32];   // K-slice of ci block
    __shared__ __align__(16) _Float16 Bsl[128][32];   // K-slice of cj block
    __shared__ float partial[8];

    const int s = blockIdx.x / NPAIRS;
    int p = blockIdx.x % NPAIRS;
    int ci = 0;
    while (p >= 63 - ci) { p -= 63 - ci; ++ci; }      // uniform scalar decode
    const int cj = ci + 1 + p;

    const _Float16* X    = s ? XbT : XaT;
    const _Float16* Arow = X + (size_t)ci * DD * KTOT;
    const _Float16* Brow = X + (size_t)cj * DD * KTOT;

    const int tid  = threadIdx.x;
    const int w    = tid >> 5;
    const int lane = tid & 31;
    const int m0   = (w & 3) * 32;       // wave M origin (4 waves along M)
    const int n0   = (w >> 2) * 64;      // wave N origin (2 waves along N)
    const int lr   = lane & 15;
    const int hi16 = lane >> 4;

    v8f accv[2][4];
    #pragma unroll
    for (int mt = 0; mt < 2; ++mt)
        #pragma unroll
        for (int nt = 0; nt < 4; ++nt)
            accv[mt][nt] = (v8f){0.f, 0.f, 0.f, 0.f, 0.f, 0.f, 0.f, 0.f};

    const int f  = tid >> 1;             // staging: feature row 0..127
    const int ko = (tid & 1) * 16;       // staging: k sub-offset 0/16

    for (int k0 = 0; k0 < KTOT; k0 += 32) {
#if USE_ASYNC_LDS
        __syncthreads();                 // previous iteration's readers done
        // 4 x 16B async copies per thread: whole 16KB K-slice, no VGPR traffic
        async_copy16(Arow + (size_t)f * KTOT + k0 + ko,     &Asl[f][ko]);
        async_copy16(Arow + (size_t)f * KTOT + k0 + ko + 8, &Asl[f][ko + 8]);
        async_copy16(Brow + (size_t)f * KTOT + k0 + ko,     &Bsl[f][ko]);
        async_copy16(Brow + (size_t)f * KTOT + k0 + ko + 8, &Bsl[f][ko + 8]);
        wait_async0();                   // this wave's copies landed in LDS
        __syncthreads();                 // all waves' copies landed
#else
        // global -> regs (coalesced b128), then regs -> LDS
        const uint4* ga = (const uint4*)(Arow + (size_t)f * KTOT + k0 + ko);
        const uint4* gb = (const uint4*)(Brow + (size_t)f * KTOT + k0 + ko);
        uint4 a0 = ga[0], a1 = ga[1];
        uint4 bq0 = gb[0], bq1 = gb[1];
        __syncthreads();                 // previous iteration's readers done
        ((uint4*)&Asl[f][ko])[0] = a0;  ((uint4*)&Asl[f][ko])[1] = a1;
        ((uint4*)&Bsl[f][ko])[0] = bq0; ((uint4*)&Bsl[f][ko])[1] = bq1;
        __syncthreads();
#endif

        // A fragment, ISA 16-bit 16x32 layout: lanes 0-15 hold K {0..7,16..23}
        v16h afrag[2], bfrag[4];
        #pragma unroll
        for (int mt = 0; mt < 2; ++mt) {
            const int mrow = m0 + mt * 16 + lr;
            const int kb   = hi16 * 8;
            h8 lo = *(const h8*)&Asl[mrow][kb];
            h8 hh = *(const h8*)&Asl[mrow][kb + 16];
            afrag[mt] = __builtin_shufflevector(lo, hh,
                0,1,2,3,4,5,6,7,8,9,10,11,12,13,14,15);
        }
        // B fragment, 32x16: lanes 0-15 hold col n, K {0..7,16..23}
        #pragma unroll
        for (int nt = 0; nt < 4; ++nt) {
            const int ncol = n0 + nt * 16 + lr;
            const int kb   = hi16 * 8;
            h8 lo = *(const h8*)&Bsl[ncol][kb];
            h8 hh = *(const h8*)&Bsl[ncol][kb + 16];
            bfrag[nt] = __builtin_shufflevector(lo, hh,
                0,1,2,3,4,5,6,7,8,9,10,11,12,13,14,15);
        }

        #pragma unroll
        for (int mt = 0; mt < 2; ++mt)
            #pragma unroll
            for (int nt = 0; nt < 4; ++nt)
                accv[mt][nt] = __builtin_amdgcn_wmma_f32_16x16x32_f16(
                    false, afrag[mt], false, bfrag[nt],
                    (short)0, accv[mt][nt], false, false);
    }

    // fused sqrt + reduce: raw v_sqrt_f32 (inputs are non-negative f32 sums;
    // 1-ulp HW sqrt avoids the 15-op IEEE expansion seen in round 0 disasm).
    // G is QSCALE^2-scaled; undo with 1/QSCALE after sqrt.
    float loc = 0.0f;
    #pragma unroll
    for (int mt = 0; mt < 2; ++mt)
        #pragma unroll
        for (int nt = 0; nt < 4; ++nt)
            #pragma unroll
            for (int j = 0; j < 8; ++j)
                loc += __builtin_amdgcn_sqrtf(accv[mt][nt][j]);
    #pragma unroll
    for (int o = 16; o; o >>= 1) loc += __shfl_xor(loc, o, 32);
    if (lane == 0) partial[w] = loc;
    __syncthreads();
    if (tid == 0) {
        float t = 0.0f;
        #pragma unroll
        for (int i = 0; i < 8; ++i) t += partial[i];
        atomicAdd(&acc[1], t * (1.0f / QSCALE));
    }
}

__global__ void eccl_finalize_kernel(const float* __restrict__ acc,
                                     float* __restrict__ out)
{
    // acc[0] = sum qa.qb ; acc[1] = sum_{s} sum_{i<j} sum sqrt(G_s block)
    // raw_a + raw_b (all i!=j) = 2 * acc[1] by block symmetry
    const float self_sim = -acc[0] / (float)(BB * NC);
    const float scale    = (float)(NC * NC - NC) * __builtin_amdgcn_sqrtf((float)BB);
    const float cluster  = -acc[1] / scale;      // -(2*acc1)/(2*scale)
    out[0] = self_sim + cluster;
}

extern "C" void kernel_launch(void* const* d_in, const int* in_sizes, int n_in,
                              void* d_out, int out_size, void* d_ws, size_t ws_size,
                              hipStream_t stream) {
    (void)in_sizes; (void)n_in; (void)out_size; (void)ws_size;
    const float* emb = (const float*)d_in[0];
    float* out = (float*)d_out;

    // workspace layout: [2 float accumulators | pad to 256B | XaT | XbT]
    float* acc = (float*)d_ws;
    _Float16* XaT = (_Float16*)((char*)d_ws + 256);
    _Float16* XbT = XaT + (size_t)NC * DD * KTOT;   // 8 MB each

    eccl_zero_kernel<<<1, 32, 0, stream>>>(acc);
    eccl_softmax_kernel<<<NC * (BB / 8), 256, 0, stream>>>(emb, XaT, XbT, acc);
    eccl_gram_kernel<<<2 * NPAIRS, 256, 0, stream>>>(XaT, XbT, acc);
    eccl_finalize_kernel<<<1, 1, 0, stream>>>(acc, out);
}
